// TGN_37460704755811
// MI455X (gfx1250) — matrix operand order repaired and checked by reading, not compiled, not run on hardware
//
#include <hip/hip_runtime.h>

typedef __attribute__((ext_vector_type(8)))  __bf16 v8bf;
typedef __attribute__((ext_vector_type(16))) __bf16 v16bf;
typedef __attribute__((ext_vector_type(8)))  float  v8f;

#define NN_NODES 100000
#define NN_EDGES 1600000
#define NODE_DIM 128
#define EDGE_DIM 64
#define HID 64
#define NPOSTS 10000

static __device__ __forceinline__ unsigned short f2bf(float f) {
  union { float f; unsigned int u; } x; x.f = f;
  unsigned int u = x.u;
  unsigned int r = u + 0x7FFFu + ((u >> 16) & 1u);   // round-to-nearest-even
  return (unsigned short)(r >> 16);
}

static __device__ __forceinline__ v16bf cat16(v8bf lo, v8bf hi) {
  return __builtin_shufflevector(lo, hi, 0,1,2,3,4,5,6,7,8,9,10,11,12,13,14,15);
}

// A operand (16x32 bf16 chunk) from LDS row-major bf16 tile, pitch in elements.
static __device__ __forceinline__ v16bf ldsA(const unsigned short* lds, int row,
                                             int pitch, int kbase, int hi) {
  const int b = row * pitch + kbase + hi * 8;
  v8bf lo = *(const v8bf*)(lds + b);
  v8bf h2 = *(const v8bf*)(lds + b + 16);
  return cat16(lo, h2);
}

// B operand (32x16 bf16 chunk): lane holds output column (= weight row `col`).
static __device__ __forceinline__ v16bf gblB(const unsigned short* w, int col,
                                             int rowlen, int kbase, int hi) {
  const int b = col * rowlen + kbase + hi * 8;
  v8bf lo = *(const v8bf*)(w + b);
  v8bf h2 = *(const v8bf*)(w + b + 16);
  return cat16(lo, h2);
}

static __device__ __forceinline__ v8f splat8(float x) {
  v8f v;
#pragma unroll
  for (int i = 0; i < 8; ++i) v[i] = x;
  return v;
}

#define WMMA_BF16(a, b, c) \
  __builtin_amdgcn_wmma_f32_16x16x32_bf16(false, (a), false, (b), (short)0, (c), false, false)

// Async global->LDS 128-bit copy (tracked by ASYNCcnt).
// ldsAddr: workgroup-relative LDS byte address (low 32 bits of generic pointer).
static __device__ __forceinline__ void async_ld_b128(unsigned ldsAddr,
                                                     unsigned long long gaddr) {
  asm volatile("global_load_async_to_lds_b128 %0, %1, off"
               :: "v"(ldsAddr), "v"(gaddr) : "memory");
}
static __device__ __forceinline__ void wait_asynccnt0() {
  asm volatile("s_wait_asynccnt 0x0" ::: "memory");
}

// saddr-form fp32 atomic add: uniform base in SGPRs + 32-bit per-lane byte offset.
static __device__ __forceinline__ void atomic_add_saddr(float* base, unsigned byteOfs,
                                                        float val) {
  asm volatile("global_atomic_add_f32 %0, %1, %2 scope:SCOPE_DEV"
               :: "v"(byteOfs), "v"(val), "s"(base) : "memory");
}

// ---------------- weight conversion ----------------
__global__ void cvt_bf16_kernel(const float* __restrict__ src,
                                unsigned short* __restrict__ dst, int n) {
  int i = blockIdx.x * blockDim.x + threadIdx.x;
  if (i < n) dst[i] = f2bf(src[i]);
}

// ---------------- encoder: out = relu(X @ W^T + b), X:[*,K], W:[64,K] ----------------
template <int K, int WAVES, bool WRITE_F32>
__global__ __launch_bounds__(256) void encode_kernel(
    const float* __restrict__ X, const unsigned short* __restrict__ Wb,
    const float* __restrict__ bias, float* __restrict__ outF,
    unsigned short* __restrict__ outB) {
  constexpr int PITCH = K + 8;
  __shared__ unsigned short lds[WAVES * 16 * PITCH];
  const int lane = threadIdx.x & 31;
  const int wave = threadIdx.x >> 5;
  const int tile = blockIdx.x * WAVES + wave;
  unsigned short* my = lds + wave * 16 * PITCH;
  const int rowBase = tile * 16;

  const float4* src = (const float4*)(X + (size_t)rowBase * K);
  constexpr int NV4 = (16 * K) / (4 * 32);
#pragma unroll
  for (int it = 0; it < NV4; ++it) {
    int e4 = it * 32 + lane;
    int e = e4 * 4;
    int r = e / K, c = e % K;
    float4 v = src[e4];
    *(unsigned int*)(my + r * PITCH + c)     = (unsigned)f2bf(v.x) | ((unsigned)f2bf(v.y) << 16);
    *(unsigned int*)(my + r * PITCH + c + 2) = (unsigned)f2bf(v.z) | ((unsigned)f2bf(v.w) << 16);
  }
  __syncthreads();

  const int hi = lane >> 4, colq = lane & 15, rowA = lane & 15;
#pragma unroll
  for (int nt = 0; nt < 4; ++nt) {
    v8f acc = splat8(bias[nt * 16 + colq]);
#pragma unroll
    for (int kc = 0; kc < K / 32; ++kc) {
      v16bf a = ldsA(my, rowA, PITCH, kc * 32, hi);
      v16bf b = gblB(Wb, nt * 16 + colq, K, kc * 32, hi);
      acc = WMMA_BF16(a, b, acc);
    }
#pragma unroll
    for (int r = 0; r < 8; ++r) {
      float v = acc[r];
      v = v > 0.f ? v : 0.f;
      int m = r + hi * 8;
      unsigned o = (unsigned)(rowBase + m) * HID + (unsigned)(nt * 16 + colq);
      if constexpr (WRITE_F32) outF[o] = v;
      outB[o] = f2bf(v);
    }
  }
}

// ---------------- edge MP: msg = [h[src]|e] @ Wagg^T + b; scatter-add to msgs[dst] ----------------
// 2 waves/block, 4 tiles (64 edges) per wave; B operands held in registers across tiles;
// activation staging via async global->LDS; segment-sum via saddr-form atomic_add_f32.
__global__ __launch_bounds__(64) void edge_mp_kernel(
    const int* __restrict__ srcIdx, const int* __restrict__ dstIdx,
    const unsigned short* __restrict__ hB, const unsigned short* __restrict__ eB,
    const unsigned short* __restrict__ Wagg, const float* __restrict__ bAgg,
    float* __restrict__ msgs) {
  constexpr int WAVES = 2, T = 4;
  constexpr int PITCH = 136;  // 128 + 8 pad (conflict-free ds_load_b128)
  __shared__ unsigned short lds[WAVES * T * 16 * PITCH];
  const int lane = threadIdx.x & 31;
  const int wave = threadIdx.x >> 5;
  unsigned short* my = lds + wave * T * 16 * PITCH;
  const int eb0 = (blockIdx.x * WAVES + wave) * T * 16;

  int sIdx[T], dIdx[T];
#pragma unroll
  for (int t = 0; t < T; ++t) {
    sIdx[t] = 0; dIdx[t] = 0;
    if (lane < 16) {
      sIdx[t] = srcIdx[eb0 + t * 16 + lane];
      dIdx[t] = dstIdx[eb0 + t * 16 + lane];
    }
  }

  // gather h_bf16[src] rows into cols 0..63, stream e_bf16 rows into cols 64..127
  const int row = lane >> 1, half = lane & 1;
#pragma unroll
  for (int t = 0; t < T; ++t) {
    int sRow = __shfl(sIdx[t], row);
    unsigned long long ha =
        (unsigned long long)(size_t)(hB + (size_t)sRow * HID + half * 32);
    unsigned long long ea =
        (unsigned long long)(size_t)(eB + (size_t)(eb0 + t * 16 + row) * HID + half * 32);
    unsigned lbase = (unsigned)(size_t)(my + (t * 16 + row) * PITCH + half * 32);
#pragma unroll
    for (int q = 0; q < 4; ++q) {
      async_ld_b128(lbase + q * 16, ha + q * 16);          // h -> cols 0..63
      async_ld_b128(lbase + 128 + q * 16, ea + q * 16);    // e -> cols 64..127
    }
  }

  // destination row byte-offsets (d * HID * 4), once per tile
  const int hi = lane >> 4, colq = lane & 15, rowA = lane & 15;
  unsigned dofsB[T][8];
#pragma unroll
  for (int t = 0; t < T; ++t)
#pragma unroll
    for (int r = 0; r < 8; ++r)
      dofsB[t][r] = (unsigned)__shfl(dIdx[t], r + hi * 8) * (unsigned)(HID * 4);

  wait_asynccnt0();
  __syncthreads();

#pragma unroll
  for (int nt = 0; nt < 4; ++nt) {
    const int col = nt * 16 + colq;
    const unsigned colB = (unsigned)col * 4u;
    v16bf B0 = gblB(Wagg, col, 128, 0, hi);
    v16bf B1 = gblB(Wagg, col, 128, 32, hi);
    v16bf B2 = gblB(Wagg, col, 128, 64, hi);
    v16bf B3 = gblB(Wagg, col, 128, 96, hi);
    const float bv = bAgg[col];
#pragma unroll
    for (int t = 0; t < T; ++t) {
      const unsigned short* tp = my + t * 16 * PITCH;
      v8f acc = splat8(bv);
      acc = WMMA_BF16(ldsA(tp, rowA, PITCH, 0,  hi), B0, acc);
      acc = WMMA_BF16(ldsA(tp, rowA, PITCH, 32, hi), B1, acc);
      acc = WMMA_BF16(ldsA(tp, rowA, PITCH, 64, hi), B2, acc);
      acc = WMMA_BF16(ldsA(tp, rowA, PITCH, 96, hi), B3, acc);
#pragma unroll
      for (int r = 0; r < 8; ++r)
        atomic_add_saddr((float*)msgs, dofsB[t][r] + colB, acc[r]);
    }
  }
}

// ---------------- GRU cell: hOut = GRU(msgs, hIn) ----------------
__global__ __launch_bounds__(64) void gru_kernel(
    const float* __restrict__ msgs, const float* __restrict__ hIn,
    const unsigned short* __restrict__ Wih, const unsigned short* __restrict__ Whh,
    const float* __restrict__ bih, const float* __restrict__ bhh,
    float* __restrict__ hOutF, unsigned short* __restrict__ hOutB) {
  constexpr int WAVES = 2;
  constexpr int PITCH = 72;
  __shared__ unsigned short lds[WAVES * 2 * 16 * PITCH];
  const int lane = threadIdx.x & 31, wave = threadIdx.x >> 5;
  const int tile = blockIdx.x * WAVES + wave;
  unsigned short* mx = lds + wave * 2 * 16 * PITCH;  // msgs tile bf16
  unsigned short* mh = mx + 16 * PITCH;              // h tile bf16
  const int nb = tile * 16;

  const float4* ms = (const float4*)(msgs + (size_t)nb * HID);
  const float4* hs = (const float4*)(hIn + (size_t)nb * HID);
#pragma unroll
  for (int it = 0; it < 8; ++it) {
    int e4 = it * 32 + lane;
    int e = e4 * 4, r = e >> 6, c = e & 63;
    float4 a = ms[e4];
    float4 b = hs[e4];
    *(unsigned int*)(mx + r * PITCH + c)     = (unsigned)f2bf(a.x) | ((unsigned)f2bf(a.y) << 16);
    *(unsigned int*)(mx + r * PITCH + c + 2) = (unsigned)f2bf(a.z) | ((unsigned)f2bf(a.w) << 16);
    *(unsigned int*)(mh + r * PITCH + c)     = (unsigned)f2bf(b.x) | ((unsigned)f2bf(b.y) << 16);
    *(unsigned int*)(mh + r * PITCH + c + 2) = (unsigned)f2bf(b.z) | ((unsigned)f2bf(b.w) << 16);
  }
  __syncthreads();

  const int hi = lane >> 4, colq = lane & 15, rowA = lane & 15;
  v16bf aM0 = ldsA(mx, rowA, PITCH, 0, hi), aM1 = ldsA(mx, rowA, PITCH, 32, hi);
  v16bf aH0 = ldsA(mh, rowA, PITCH, 0, hi), aH1 = ldsA(mh, rowA, PITCH, 32, hi);

#pragma unroll
  for (int nt = 0; nt < 4; ++nt) {
    const int col = nt * 16 + colq;
    v8f ir = splat8(bih[col]), iz = splat8(bih[64 + col]), in_ = splat8(bih[128 + col]);
    v8f hr = splat8(bhh[col]), hz = splat8(bhh[64 + col]), hn  = splat8(bhh[128 + col]);

    ir  = WMMA_BF16(aM0, gblB(Wih, col,        64, 0,  hi), ir);
    ir  = WMMA_BF16(aM1, gblB(Wih, col,        64, 32, hi), ir);
    iz  = WMMA_BF16(aM0, gblB(Wih, 64 + col,   64, 0,  hi), iz);
    iz  = WMMA_BF16(aM1, gblB(Wih, 64 + col,   64, 32, hi), iz);
    in_ = WMMA_BF16(aM0, gblB(Wih, 128 + col,  64, 0,  hi), in_);
    in_ = WMMA_BF16(aM1, gblB(Wih, 128 + col,  64, 32, hi), in_);
    hr  = WMMA_BF16(aH0, gblB(Whh, col,        64, 0,  hi), hr);
    hr  = WMMA_BF16(aH1, gblB(Whh, col,        64, 32, hi), hr);
    hz  = WMMA_BF16(aH0, gblB(Whh, 64 + col,   64, 0,  hi), hz);
    hz  = WMMA_BF16(aH1, gblB(Whh, 64 + col,   64, 32, hi), hz);
    hn  = WMMA_BF16(aH0, gblB(Whh, 128 + col,  64, 0,  hi), hn);
    hn  = WMMA_BF16(aH1, gblB(Whh, 128 + col,  64, 32, hi), hn);

#pragma unroll
    for (int r = 0; r < 8; ++r) {
      int m = r + hi * 8;
      float rg = 1.f / (1.f + __expf(-(ir[r] + hr[r])));
      float zg = 1.f / (1.f + __expf(-(iz[r] + hz[r])));
      float nx = in_[r] + rg * hn[r];
      float e2 = __expf(2.f * nx);
      float ng = (e2 - 1.f) / (e2 + 1.f);  // tanh
      unsigned o = (unsigned)(nb + m) * HID + (unsigned)col;
      float hv = (1.f - zg) * ng + zg * hIn[o];
      hOutF[o] = hv;
      hOutB[o] = f2bf(hv);
    }
  }
}

// ---------------- readout: sigmoid(relu(h[mask] @ W1^T + b1) @ W2^T + b2) ----------------
__global__ __launch_bounds__(256) void readout_kernel(
    const float* __restrict__ h, const int* __restrict__ mask,
    const float* __restrict__ W1, const float* __restrict__ b1,
    const float* __restrict__ W2, const float* __restrict__ b2,
    float* __restrict__ out) {
  __shared__ float w1[32 * 64];
  __shared__ float w2[32];
  __shared__ float bb1[32];
  for (int i = threadIdx.x; i < 32 * 64; i += blockDim.x) w1[i] = W1[i];
  if (threadIdx.x < 32) { w2[threadIdx.x] = W2[threadIdx.x]; bb1[threadIdx.x] = b1[threadIdx.x]; }
  __syncthreads();
  int p = blockIdx.x * blockDim.x + threadIdx.x;
  if (p >= NPOSTS) return;
  const float* row = h + (size_t)mask[p] * HID;
  float x[64];
#pragma unroll
  for (int k = 0; k < 64; ++k) x[k] = row[k];
  float acc = b2[0];
#pragma unroll
  for (int j = 0; j < 32; ++j) {
    float s = bb1[j];
#pragma unroll
    for (int k = 0; k < 64; ++k) s += x[k] * w1[j * 64 + k];
    s = s > 0.f ? s : 0.f;
    acc += s * w2[j];
  }
  out[p] = 1.f / (1.f + __expf(-acc));
}

extern "C" void kernel_launch(void* const* d_in, const int* in_sizes, int n_in,
                              void* d_out, int out_size, void* d_ws, size_t ws_size,
                              hipStream_t stream) {
  (void)in_sizes; (void)n_in; (void)out_size; (void)ws_size;
  const float* node_features = (const float*)d_in[0];
  const float* edge_attr     = (const float*)d_in[1];
  const int*   edge_index    = (const int*)d_in[2];
  const int*   post_mask     = (const int*)d_in[3];
  // d_in[4] = n_layers (fixed 2 in setup; launch must be deterministic)
  const float* W_node = (const float*)d_in[5];  const float* b_node = (const float*)d_in[6];
  const float* W_edge = (const float*)d_in[7];  const float* b_edge = (const float*)d_in[8];
  const float* W_agg  = (const float*)d_in[9];  const float* b_agg  = (const float*)d_in[10];
  const float* w_ih   = (const float*)d_in[11]; const float* w_hh   = (const float*)d_in[12];
  const float* b_ih   = (const float*)d_in[13]; const float* b_hh   = (const float*)d_in[14];
  const float* W_out1 = (const float*)d_in[15]; const float* b_out1 = (const float*)d_in[16];
  const float* W_out2 = (const float*)d_in[17]; const float* b_out2 = (const float*)d_in[18];

  char* ws = (char*)d_ws;
  size_t off = 0;
  auto alloc = [&](size_t bytes) {
    void* p = ws + off;
    off = (off + bytes + 255) & ~(size_t)255;
    return p;
  };
  float*          hA   = (float*)alloc((size_t)NN_NODES * HID * 4);
  float*          hBuf = (float*)alloc((size_t)NN_NODES * HID * 4);
  unsigned short* hAb  = (unsigned short*)alloc((size_t)NN_NODES * HID * 2);
  unsigned short* hBb  = (unsigned short*)alloc((size_t)NN_NODES * HID * 2);
  unsigned short* eBb  = (unsigned short*)alloc((size_t)NN_EDGES * HID * 2);
  float*          msgs = (float*)alloc((size_t)NN_NODES * HID * 4);
  unsigned short* WnB  = (unsigned short*)alloc((size_t)HID * NODE_DIM * 2);
  unsigned short* WeB  = (unsigned short*)alloc((size_t)HID * EDGE_DIM * 2);
  unsigned short* WaB  = (unsigned short*)alloc((size_t)HID * 2 * HID * 2);
  unsigned short* WihB = (unsigned short*)alloc((size_t)3 * HID * HID * 2);
  unsigned short* WhhB = (unsigned short*)alloc((size_t)3 * HID * HID * 2);

  // 1) weights -> bf16
  cvt_bf16_kernel<<<(HID * NODE_DIM + 255) / 256, 256, 0, stream>>>(W_node, WnB, HID * NODE_DIM);
  cvt_bf16_kernel<<<(HID * EDGE_DIM + 255) / 256, 256, 0, stream>>>(W_edge, WeB, HID * EDGE_DIM);
  cvt_bf16_kernel<<<(HID * 2 * HID + 255) / 256, 256, 0, stream>>>(W_agg, WaB, HID * 2 * HID);
  cvt_bf16_kernel<<<(3 * HID * HID + 255) / 256, 256, 0, stream>>>(w_ih, WihB, 3 * HID * HID);
  cvt_bf16_kernel<<<(3 * HID * HID + 255) / 256, 256, 0, stream>>>(w_hh, WhhB, 3 * HID * HID);

  // 2) encoders (WMMA bf16, fp32 accum)
  encode_kernel<NODE_DIM, 2, true><<<NN_NODES / 16 / 2, 64, 0, stream>>>(
      node_features, WnB, b_node, hA, hAb);
  encode_kernel<EDGE_DIM, 8, false><<<NN_EDGES / 16 / 8, 256, 0, stream>>>(
      edge_attr, WeB, b_edge, nullptr, eBb);

  const int* srcIdx = edge_index;
  const int* dstIdx = edge_index + NN_EDGES;

  // 3) message-passing layers (n_layers == 2)
  for (int L = 0; L < 2; ++L) {
    const float* curF = (L == 0) ? hA : hBuf;
    const unsigned short* curB = (L == 0) ? hAb : hBb;
    float* nxtF = (L == 0) ? hBuf : hA;
    unsigned short* nxtB = (L == 0) ? hBb : hAb;

    hipMemsetAsync(msgs, 0, (size_t)NN_NODES * HID * 4, stream);
    edge_mp_kernel<<<NN_EDGES / 16 / (2 * 4), 64, 0, stream>>>(
        srcIdx, dstIdx, curB, eBb, WaB, b_agg, msgs);
    gru_kernel<<<NN_NODES / 16 / 2, 64, 0, stream>>>(
        msgs, curF, WihB, WhhB, b_ih, b_hh, nxtF, nxtB);
  }

  // 4) readout (h back in hA after 2 layers)
  readout_kernel<<<(NPOSTS + 255) / 256, 256, 0, stream>>>(
      hA, post_mask, W_out1, b_out1, W_out2, b_out2, (float*)d_out);
}
